// AttentionFold_58669253263948
// MI455X (gfx1250) — matrix operand order repaired and checked by reading, not compiled, not run on hardware
//
#include <hip/hip_runtime.h>
#include <hip/hip_bf16.h>

// ---------------------------------------------------------------------------
// CDNA5 (gfx1250) fused AttentionFold.
// bf16 WMMA (v_wmma_f32_16x16x32_bf16) for all GEMMs, f32 softmax/sigmoid.
// ---------------------------------------------------------------------------

typedef __attribute__((ext_vector_type(16))) __bf16 v16bf;
typedef __attribute__((ext_vector_type(8)))  float  v8f;

union UA { v16bf v; __bf16 e[16]; };
union UC { v8f  v; float  e[8];  };

#define NB 16      // batches
#define PP 4096    // points per batch
#define CC 128     // channels
#define QQ 784     // H*W
#define QP 800     // padded H*W (25 * 32)
#define GG 512     // glob dim
#define MROWS 32   // rows per workgroup

// A fragment: 16x32 bf16 tile, row-major source with stride lda.
// Layout used here (contiguous k): lane<16 -> M=lane, e0..15=K0..15
//             lane>=16 -> M=lane-16, e0..15=K16..31
__device__ __forceinline__ v16bf ldA(const __bf16* A, int lda, int row0, int k0, int lane) {
  const int half = lane >> 4, m = lane & 15;
  const __bf16* p = A + (row0 + m) * lda + k0 + half * 16;
  UA u;
  #pragma unroll
  for (int i = 0; i < 8; ++i) { u.e[i] = p[i]; u.e[8 + i] = p[8 + i]; }
  return u.v;
}

// B fragment: 32x16 bf16 tile; B stored column-major ([N][K] row-major, stride ldb)
// Layout used here (same k order as ldA): lane holds col N=lane%16, K = k0 + 16*(lane/16) + e, e=0..15 contiguous.
__device__ __forceinline__ v16bf ldB(const __bf16* B, int ldb, int n0, int k0, int lane) {
  const __bf16* p = B + (size_t)(n0 + (lane & 15)) * ldb + k0 + (lane >> 4) * 16;
  UA u;
  #pragma unroll
  for (int i = 0; i < 16; ++i) u.e[i] = p[i];
  return u.v;
}

#define WMMA_BF16(accv, av, bv) \
  (accv) = __builtin_amdgcn_wmma_f32_16x16x32_bf16(false, (av), false, (bv), (short)0, (accv), false, false)

// ---------------------------------------------------------------------------
// Kernel 0: weight transpose + bf16 convert + K-packing into [N][K] layouts.
// ---------------------------------------------------------------------------
__global__ void __launch_bounds__(256)
wcvt_kernel(const float* __restrict__ sw1, const float* __restrict__ gw1,
            const float* __restrict__ sw2, const float* __restrict__ gw2,
            const float* __restrict__ fw1, const float* __restrict__ fw2,
            __bf16* __restrict__ sw1T, __bf16* __restrict__ gw1T,
            __bf16* __restrict__ sw2T, __bf16* __restrict__ gw2T,
            __bf16* __restrict__ fw1T, __bf16* __restrict__ fw2T) {
  int idx = blockIdx.x * 256 + threadIdx.x;
  const int S0 = 128 * 32, S1 = 128 * 32, S2 = QQ * 128, S3 = 128 * 128,
            S4 = 128 * 160, S5 = 128 * 128;
  if (idx < S0) { int j = idx >> 5, k = idx & 31;
    sw1T[idx] = (__bf16)(k < 15 ? sw1[k * 128 + j] : 0.f); return; }
  idx -= S0;
  if (idx < S1) { int j = idx >> 5, k = idx & 31;
    gw1T[idx] = (__bf16)(k < 15 ? gw1[k * 128 + j] : 0.f); return; }
  idx -= S1;
  if (idx < S2) { int q = idx >> 7, k = idx & 127;
    sw2T[idx] = (__bf16)sw2[k * QQ + q]; return; }
  idx -= S2;
  if (idx < S3) { int j = idx >> 7, k = idx & 127;
    gw2T[idx] = (__bf16)gw2[k * 128 + j]; return; }
  idx -= S3;
  if (idx < S4) {
    int j = idx / 160, k = idx - j * 160;
    float v = 0.f;
    if (k < 15)       v = fw1[k * 128 + j];              // points+transform rows
    else if (k < 17)  v = fw1[(527 + (k - 15)) * 128 + j]; // coords rows
    else if (k < 145) v = fw1[(529 + (k - 17)) * 128 + j]; // spat_feats rows
    fw1T[idx] = (__bf16)v; return;
  }
  idx -= S4;
  if (idx < S5) { int j = idx >> 7, k = idx & 127;
    fw2T[idx] = (__bf16)fw2[k * 128 + j]; }
}

// ---------------------------------------------------------------------------
// Kernel 1: enc_filters (N,C,Q) f32 -> bf16 padded to Q=800 (zeros), [N][C][QP].
// This is exactly the [N][K] layout needed for the attention GEMM B matrix.
// ---------------------------------------------------------------------------
__global__ void __launch_bounds__(256)
fcvt_kernel(const float* __restrict__ ef, __bf16* __restrict__ fb) {
  size_t idx = (size_t)blockIdx.x * 256 + threadIdx.x;
  if (idx >= (size_t)NB * CC * QP) return;
  int q = (int)(idx % QP);
  size_t nc = idx / QP;
  fb[idx] = (__bf16)(q < QQ ? ef[nc * QQ + q] : 0.f);
}

// ---------------------------------------------------------------------------
// Kernel 2: per-batch glob bases:
//   g_base[n][j] = glob[n] @ gw1[15:527, j] + gb1[j]
//   f_base[n][j] = glob[n] @ fw1[15:527, j] + fb1[j]
// ---------------------------------------------------------------------------
__global__ void __launch_bounds__(128)
base_kernel(const float* __restrict__ glob,
            const float* __restrict__ gw1, const float* __restrict__ gb1,
            const float* __restrict__ fw1, const float* __restrict__ fb1,
            float* __restrict__ g_base, float* __restrict__ f_base) {
  int n = blockIdx.x, j = threadIdx.x;
  float ag = gb1[j], af = fb1[j];
  for (int k = 0; k < GG; ++k) {
    float g = glob[n * GG + k];
    ag += g * gw1[(15 + k) * 128 + j];
    af += g * fw1[(15 + k) * 128 + j];
  }
  g_base[n * 128 + j] = ag;
  f_base[n * 128 + j] = af;
}

// ---------------------------------------------------------------------------
// Kernel 3: fused main pipeline. 1 workgroup = 32 points of one batch,
// 8 waves, ~202 KB dynamic LDS. All GEMMs via v_wmma_f32_16x16x32_bf16.
// ---------------------------------------------------------------------------
#define SMEM_BYTES 206848

__global__ void __launch_bounds__(256)
fold_main(const float* __restrict__ points, const float* __restrict__ transform,
          const float* __restrict__ sb1, const float* __restrict__ sb2,
          const float* __restrict__ gb2, const float* __restrict__ fb2,
          const float* __restrict__ fw3, const float* __restrict__ fb3,
          const __bf16* __restrict__ sw1T, const __bf16* __restrict__ gw1T,
          const __bf16* __restrict__ sw2T, const __bf16* __restrict__ gw2T,
          const __bf16* __restrict__ fw1T, const __bf16* __restrict__ fw2T,
          const __bf16* __restrict__ filtB,
          const float* __restrict__ g_base, const float* __restrict__ f_base,
          float* __restrict__ out_pre) {
  extern __shared__ char smem[];
  __bf16* A15 = (__bf16*)(smem);            // 32x32  bf16 (pt15 zero-padded)
  __bf16* SH  = (__bf16*)(smem + 2048);     // 32x128 bf16
  __bf16* GH  = (__bf16*)(smem + 10240);    // 32x128 bf16
  __bf16* GAT = (__bf16*)(smem + 18432);    // 32x128 bf16 sigmoid gate
  __bf16* A2  = (__bf16*)(smem + 26624);    // 32x160 bf16 packed fold input
  __bf16* H1  = (__bf16*)(smem + 36864);    // 32x128 bf16
  __bf16* H2  = (__bf16*)(smem + 45056);    // 32x128 bf16
  float*  LG  = (float*)(smem + 53248);     // 32x800 f32 logits
  __bf16* PR  = (__bf16*)(smem + 155648);   // 32x800 bf16 softmax probs

  const int tid = threadIdx.x, lane = tid & 31, wave = tid >> 5;
  const int n  = blockIdx.x >> 7;
  const int p0 = (blockIdx.x & 127) * MROWS;

  // ---- stage 0: fill A15 (spat/gate input) and A2 base columns ----
  for (int t = tid; t < MROWS * 32; t += 256) {
    int r = t >> 5, k = t & 31, p = p0 + r;
    float v = 0.f;
    if (k < 3)       v = points[(size_t)(n * PP + p) * 3 + k];
    else if (k < 15) v = transform[(size_t)(n * PP + p) * 12 + (k - 3)];
    A15[t] = (__bf16)v;
  }
  for (int t = tid; t < MROWS * 160; t += 256) {
    int r = t / 160, k = t - r * 160, p = p0 + r;
    float v = 0.f;
    if (k < 3)        v = points[(size_t)(n * PP + p) * 3 + k];
    else if (k < 15)  v = transform[(size_t)(n * PP + p) * 12 + (k - 3)];
    else if (k == 15) v = -1.f + 2.f * (float)(p >> 6) / 63.f;   // grid x
    else if (k == 16) v = -1.f + 2.f * (float)(p & 63) / 63.f;   // grid y
    A2[t] = (__bf16)v;   // cols 17..144 overwritten by attention stage
  }
  __syncthreads();

  // ---- stage 1: sh = relu(pt15@sw1+sb1), gh = relu(pt15@gw1+g_base) ----
  for (int t = wave; t < 32; t += 8) {
    const bool isS = t < 16;
    const int u = isS ? t : t - 16;
    const int mt = u >> 3, nt = u & 7;
    UC acc;
    #pragma unroll
    for (int i = 0; i < 8; ++i) acc.e[i] = 0.f;
    v16bf a = ldA(A15, 32, mt * 16, 0, lane);
    v16bf b = ldB(isS ? sw1T : gw1T, 32, nt * 16, 0, lane);
    WMMA_BF16(acc.v, a, b);
    const int col = nt * 16 + (lane & 15);
    const float bias = isS ? sb1[col] : g_base[n * 128 + col];
    __bf16* dst = isS ? SH : GH;
    #pragma unroll
    for (int r = 0; r < 8; ++r) {
      int row = mt * 16 + r + ((lane >> 4) << 3);
      dst[row * 128 + col] = (__bf16)fmaxf(acc.e[r] + bias, 0.f);
    }
  }
  __syncthreads();

  // ---- stage 2: logits = sh@sw2+sb2 (f32), gattn = sigmoid(gh@gw2+gb2) ----
  for (int t = wave; t < 114; t += 8) {
    if (t < 98) {
      const int mt = t / 49, nt = t - mt * 49;
      UC acc;
      #pragma unroll
      for (int i = 0; i < 8; ++i) acc.e[i] = 0.f;
      #pragma unroll
      for (int ks = 0; ks < 4; ++ks) {
        v16bf a = ldA(SH, 128, mt * 16, ks * 32, lane);
        v16bf b = ldB(sw2T, 128, nt * 16, ks * 32, lane);
        WMMA_BF16(acc.v, a, b);
      }
      const int col = nt * 16 + (lane & 15);
      const float bias = sb2[col];
      #pragma unroll
      for (int r = 0; r < 8; ++r) {
        int row = mt * 16 + r + ((lane >> 4) << 3);
        LG[row * QP + col] = acc.e[r] + bias;
      }
    } else {
      const int u = t - 98;
      const int mt = u >> 3, nt = u & 7;
      UC acc;
      #pragma unroll
      for (int i = 0; i < 8; ++i) acc.e[i] = 0.f;
      #pragma unroll
      for (int ks = 0; ks < 4; ++ks) {
        v16bf a = ldA(GH, 128, mt * 16, ks * 32, lane);
        v16bf b = ldB(gw2T, 128, nt * 16, ks * 32, lane);
        WMMA_BF16(acc.v, a, b);
      }
      const int col = nt * 16 + (lane & 15);
      const float bias = gb2[col];
      #pragma unroll
      for (int r = 0; r < 8; ++r) {
        int row = mt * 16 + r + ((lane >> 4) << 3);
        float x = acc.e[r] + bias;
        GAT[row * 128 + col] = (__bf16)(1.f / (1.f + __expf(-x)));
      }
    }
  }
  __syncthreads();

  // ---- stage 3: row softmax over 784 (f32), write bf16 probs, zero pad ----
  for (int rr = 0; rr < 4; ++rr) {
    const int row = wave * 4 + rr;
    float v[25];
    float mx = -1e30f;
    #pragma unroll
    for (int j = 0; j < 25; ++j) {
      int i = lane + 32 * j;
      float x = (i < QQ) ? LG[row * QP + i] : -1e30f;
      v[j] = x;
      mx = fmaxf(mx, x);
    }
    #pragma unroll
    for (int off = 16; off >= 1; off >>= 1) mx = fmaxf(mx, __shfl_xor(mx, off, 32));
    float s = 0.f;
    #pragma unroll
    for (int j = 0; j < 25; ++j) {
      float e = (v[j] > -1e29f) ? __expf(v[j] - mx) : 0.f;
      v[j] = e; s += e;
    }
    #pragma unroll
    for (int off = 16; off >= 1; off >>= 1) s += __shfl_xor(s, off, 32);
    const float inv = 1.f / s;
    #pragma unroll
    for (int j = 0; j < 25; ++j) {
      int i = lane + 32 * j;
      PR[row * QP + i] = (__bf16)(v[j] * inv);
    }
  }
  __syncthreads();

  // ---- stage 4: spat_feats = gattn * (probs @ filters^T); pack into A2 ----
  {
    const __bf16* FB = filtB + (size_t)n * CC * QP;
    for (int t = wave; t < 16; t += 8) {
      const int mt = t >> 3, nt = t & 7;
      UC acc;
      #pragma unroll
      for (int i = 0; i < 8; ++i) acc.e[i] = 0.f;
      for (int ks = 0; ks < 25; ++ks) {
        v16bf a = ldA(PR, QP, mt * 16, ks * 32, lane);
        v16bf b = ldB(FB, QP, nt * 16, ks * 32, lane);
        WMMA_BF16(acc.v, a, b);
      }
      const int col = nt * 16 + (lane & 15);
      #pragma unroll
      for (int r = 0; r < 8; ++r) {
        int row = mt * 16 + r + ((lane >> 4) << 3);
        float g = (float)GAT[row * 128 + col];
        A2[row * 160 + 17 + col] = (__bf16)(acc.e[r] * g);
      }
    }
  }
  __syncthreads();

  // ---- stage 5: h1 = relu(A2 @ fw1p + f_base) ----
  for (int t = wave; t < 16; t += 8) {
    const int mt = t >> 3, nt = t & 7;
    UC acc;
    #pragma unroll
    for (int i = 0; i < 8; ++i) acc.e[i] = 0.f;
    #pragma unroll
    for (int ks = 0; ks < 5; ++ks) {
      v16bf a = ldA(A2, 160, mt * 16, ks * 32, lane);
      v16bf b = ldB(fw1T, 160, nt * 16, ks * 32, lane);
      WMMA_BF16(acc.v, a, b);
    }
    const int col = nt * 16 + (lane & 15);
    const float bias = f_base[n * 128 + col];
    #pragma unroll
    for (int r = 0; r < 8; ++r) {
      int row = mt * 16 + r + ((lane >> 4) << 3);
      H1[row * 128 + col] = (__bf16)fmaxf(acc.e[r] + bias, 0.f);
    }
  }
  __syncthreads();

  // ---- stage 6: h2 = relu(h1 @ fw2 + fb2) ----
  for (int t = wave; t < 16; t += 8) {
    const int mt = t >> 3, nt = t & 7;
    UC acc;
    #pragma unroll
    for (int i = 0; i < 8; ++i) acc.e[i] = 0.f;
    #pragma unroll
    for (int ks = 0; ks < 4; ++ks) {
      v16bf a = ldA(H1, 128, mt * 16, ks * 32, lane);
      v16bf b = ldB(fw2T, 128, nt * 16, ks * 32, lane);
      WMMA_BF16(acc.v, a, b);
    }
    const int col = nt * 16 + (lane & 15);
    const float bias = fb2[col];
    #pragma unroll
    for (int r = 0; r < 8; ++r) {
      int row = mt * 16 + r + ((lane >> 4) << 3);
      H2[row * 128 + col] = (__bf16)fmaxf(acc.e[r] + bias, 0.f);
    }
  }
  __syncthreads();

  // ---- stage 7: deform = h2 @ fw3 + fb3 ; out_pre = deform + points ----
  if (tid < MROWS * 3) {
    const int r = tid / 3, d = tid - r * 3, p = p0 + r;
    float acc = fb3[d];
    for (int k = 0; k < 128; ++k) acc += (float)H2[r * 128 + k] * fw3[k * 3 + d];
    out_pre[(size_t)(n * PP + p) * 3 + d] = acc + points[(size_t)(n * PP + p) * 3 + d];
  }
}

// ---------------------------------------------------------------------------
// Kernel 4: per-batch normalize: center by mean, scale by max ||c|| + 1e-8.
// ---------------------------------------------------------------------------
__global__ void __launch_bounds__(256)
norm_kernel(const float* __restrict__ pre, float* __restrict__ out) {
  __shared__ float rx[256], ry[256], rz[256];
  __shared__ float mean_s[3], scale_s;
  const int n = blockIdx.x, tid = threadIdx.x;
  const float* src = pre + (size_t)n * PP * 3;
  float sx = 0.f, sy = 0.f, sz = 0.f;
  for (int p = tid; p < PP; p += 256) {
    sx += src[p * 3 + 0]; sy += src[p * 3 + 1]; sz += src[p * 3 + 2];
  }
  rx[tid] = sx; ry[tid] = sy; rz[tid] = sz;
  __syncthreads();
  for (int o = 128; o > 0; o >>= 1) {
    if (tid < o) { rx[tid] += rx[tid + o]; ry[tid] += ry[tid + o]; rz[tid] += rz[tid + o]; }
    __syncthreads();
  }
  if (tid == 0) {
    mean_s[0] = rx[0] / (float)PP; mean_s[1] = ry[0] / (float)PP; mean_s[2] = rz[0] / (float)PP;
  }
  __syncthreads();
  const float m0 = mean_s[0], m1 = mean_s[1], m2 = mean_s[2];
  float mx = 0.f;
  for (int p = tid; p < PP; p += 256) {
    float a = src[p * 3 + 0] - m0, b = src[p * 3 + 1] - m1, c = src[p * 3 + 2] - m2;
    mx = fmaxf(mx, sqrtf(a * a + b * b + c * c));
  }
  rx[tid] = mx;
  __syncthreads();
  for (int o = 128; o > 0; o >>= 1) {
    if (tid < o) rx[tid] = fmaxf(rx[tid], rx[tid + o]);
    __syncthreads();
  }
  if (tid == 0) scale_s = rx[0] + 1e-8f;
  __syncthreads();
  const float inv = 1.f / scale_s;
  float* dst = out + (size_t)n * PP * 3;
  for (int p = tid; p < PP; p += 256) {
    dst[p * 3 + 0] = (src[p * 3 + 0] - m0) * inv;
    dst[p * 3 + 1] = (src[p * 3 + 1] - m1) * inv;
    dst[p * 3 + 2] = (src[p * 3 + 2] - m2) * inv;
  }
}

// ---------------------------------------------------------------------------
extern "C" void kernel_launch(void* const* d_in, const int* in_sizes, int n_in,
                              void* d_out, int out_size, void* d_ws, size_t ws_size,
                              hipStream_t stream) {
  (void)in_sizes; (void)n_in; (void)out_size; (void)ws_size;
  const float* points    = (const float*)d_in[0];
  const float* transform = (const float*)d_in[1];
  const float* enc_filt  = (const float*)d_in[2];
  const float* enc_glob  = (const float*)d_in[3];
  const float* sw1 = (const float*)d_in[4];  const float* sb1 = (const float*)d_in[5];
  const float* sw2 = (const float*)d_in[6];  const float* sb2 = (const float*)d_in[7];
  const float* gw1 = (const float*)d_in[8];  const float* gb1 = (const float*)d_in[9];
  const float* gw2 = (const float*)d_in[10]; const float* gb2 = (const float*)d_in[11];
  const float* fw1 = (const float*)d_in[12]; const float* fb1 = (const float*)d_in[13];
  const float* fw2 = (const float*)d_in[14]; const float* fb2 = (const float*)d_in[15];
  const float* fw3 = (const float*)d_in[16]; const float* fb3 = (const float*)d_in[17];
  float* out = (float*)d_out;

  char* ws = (char*)d_ws;
  size_t off = 0;
  auto walloc = [&](size_t bytes) -> char* {
    char* p = ws + off;
    off += (bytes + 255) & ~(size_t)255;
    return p;
  };
  __bf16* sw1T   = (__bf16*)walloc((size_t)128 * 32 * 2);
  __bf16* gw1T   = (__bf16*)walloc((size_t)128 * 32 * 2);
  __bf16* sw2T   = (__bf16*)walloc((size_t)QQ * 128 * 2);
  __bf16* gw2T   = (__bf16*)walloc((size_t)128 * 128 * 2);
  __bf16* fw1T   = (__bf16*)walloc((size_t)128 * 160 * 2);
  __bf16* fw2T   = (__bf16*)walloc((size_t)128 * 128 * 2);
  __bf16* filtB  = (__bf16*)walloc((size_t)NB * CC * QP * 2);
  float*  g_base = (float*)walloc((size_t)NB * 128 * 4);
  float*  f_base = (float*)walloc((size_t)NB * 128 * 4);
  float*  outpre = (float*)walloc((size_t)NB * PP * 3 * 4);

  (void)hipFuncSetAttribute((const void*)fold_main,
                            hipFuncAttributeMaxDynamicSharedMemorySize, SMEM_BYTES);

  // 0) weight convert/transpose/pack: 161792 elements total
  wcvt_kernel<<<632, 256, 0, stream>>>(sw1, gw1, sw2, gw2, fw1, fw2,
                                       sw1T, gw1T, sw2T, gw2T, fw1T, fw2T);
  // 1) filters -> bf16 padded [N][C][800]
  fcvt_kernel<<<(NB * CC * QP) / 256, 256, 0, stream>>>(enc_filt, filtB);
  // 2) per-batch glob bases
  base_kernel<<<NB, 128, 0, stream>>>(enc_glob, gw1, gb1, fw1, fb1, g_base, f_base);
  // 3) fused main pipeline: 16 batches * 128 tiles of 32 points
  fold_main<<<NB * 128, 256, SMEM_BYTES, stream>>>(
      points, transform, sb1, sb2, gb2, fb2, fw3, fb3,
      sw1T, gw1T, sw2T, gw2T, fw1T, fw2T, filtB, g_base, f_base, outpre);
  // 4) per-batch normalization
  norm_kernel<<<NB, 256, 0, stream>>>(outpre, out);
}
